// ThermoGateLayer_30880814858966
// MI455X (gfx1250) — matrix operand order repaired and checked by reading, not compiled, not run on hardware
//
#include <hip/hip_runtime.h>
#include <hip/hip_bf16.h>
#include <math.h>

// ---------------------------------------------------------------------------
// Types for CDNA5 WMMA (wave32): D(16x16 f32) = A(16x32 bf16) x B(32x16 bf16) + C
// ---------------------------------------------------------------------------
typedef __attribute__((ext_vector_type(16))) __bf16 v16bf;
typedef __attribute__((ext_vector_type(8)))  float  v8f;
typedef __attribute__((ext_vector_type(4)))  unsigned int u32x4;
// TDM descriptor vectors (6-arg tensor_load_to_lds on this toolchain)
typedef __attribute__((ext_vector_type(4)))  unsigned int uv4i;
typedef __attribute__((ext_vector_type(8)))  int          iv8;
typedef __attribute__((ext_vector_type(4)))  int          iv4;

union BF16Frag {
    v16bf v;
    u32x4 q[2];
    unsigned short h[16];
};

__device__ __forceinline__ unsigned short f2bf_u(float f) {
    union { float f; unsigned int u; } c; c.f = f;
    unsigned int u = c.u;
    unsigned int r = u + 0x7FFFu + ((u >> 16) & 1u);   // round-to-nearest-even
    return (unsigned short)(r >> 16);
}

#define WMMA_BF16(A, B, C) \
    __builtin_amdgcn_wmma_f32_16x16x32_bf16(false, (A), false, (B), (short)0, (C), false, false)

// Problem sizes
#define BB   8
#define TT   2048
#define CC   768
#define HH   12
#define HD   64
#define DFF  3072
#define KSEL 1024
#define MT   (BB*TT)     // 16384 tokens
#define MS   (BB*KSEL)   // 8192 selected tokens
#define HOUT ((size_t)MT*CC)   // 12582912

// ---------------------------------------------------------------------------
// TDM: stage a 2D bf16 tile [rows x tile_w] (row stride = stride_elems) from
// global into LDS at byte offset lds_off. One issue per wave; TENSORcnt.
// ---------------------------------------------------------------------------
__device__ __forceinline__ void tdm_load_tile_2d(unsigned lds_off,
                                                 const unsigned short* gptr,
                                                 int rows, int tile_w,
                                                 int stride_elems) {
    unsigned long long ga = (unsigned long long)(size_t)gptr;
    uv4i g0;
    g0[0] = 1u;                                           // count=1 (valid user D#)
    g0[1] = lds_off;                                      // lds_addr (bytes)
    g0[2] = (unsigned)(ga & 0xFFFFFFFFu);                 // global_addr[31:0]
    g0[3] = (unsigned)((ga >> 32) & 0x01FFFFFFu)          // global_addr[56:32]
          | (2u << 30);                                   // type = 2 ("image")
    iv8 g1;
    g1[0] = (int)(1u << 16);                              // data_size=1 (2B), wg_mask=0
    g1[1] = (int)(((unsigned)tile_w & 0xFFFFu) << 16);    // tensor_dim0[15:0]
    g1[2] = (int)((((unsigned)tile_w >> 16) & 0xFFFFu)    // tensor_dim0[31:16]
          | (((unsigned)rows & 0xFFFFu) << 16));          // tensor_dim1[15:0]
    g1[3] = (int)((((unsigned)rows >> 16) & 0xFFFFu)      // tensor_dim1[31:16]
          | (((unsigned)tile_w & 0xFFFFu) << 16));        // tile_dim0
    g1[4] = (int)((unsigned)rows & 0xFFFFu);              // tile_dim1 ; tile_dim2=0
    g1[5] = stride_elems;                                 // tensor_dim0_stride[31:0]
    g1[6] = 0;                                            // stride hi / dim1_stride lo
    g1[7] = 0;
    iv4 z4 = {0, 0, 0, 0};
    iv8 z8 = {0, 0, 0, 0, 0, 0, 0, 0};
    __builtin_amdgcn_tensor_load_to_lds(g0, g1, z4, z4, z8, 0);
}

// ---------------------------------------------------------------------------
// f32 -> bf16 elementwise conversion
// ---------------------------------------------------------------------------
__global__ __launch_bounds__(256) void f32_to_bf16_k(const float* __restrict__ in,
                                                     unsigned short* __restrict__ out, int n) {
    int i = blockIdx.x * 256 + threadIdx.x;
    if (i < n) out[i] = f2bf_u(in[i]);
}

// ---------------------------------------------------------------------------
// Generic WMMA GEMM: out[M,N] = act(A[M,K] @ W[N,K]^T + bias) (+ res)
// Block = 256 threads = 8 waves (2 x 4). Block tile: 32(M) x 256(N).
// Weight tile [256 x 64] staged into LDS by the Tensor Data Mover, double
// buffered (2 x 32KB dynamic LDS); compute on buf k overlaps TDM of k+1.
// act: 0 = none, 1 = exact GELU
// ---------------------------------------------------------------------------
__device__ __forceinline__ void gemm_epilogue(v8f acc, int mbase, int col, int N,
                                              const float* bias, const float* res,
                                              float* outf, unsigned short* outb, int act) {
    float bv = bias ? bias[col] : 0.0f;
#pragma unroll
    for (int j = 0; j < 8; ++j) {
        float v = acc[j] + bv;
        if (act == 1) v = 0.5f * v * (1.0f + erff(v * 0.70710678118f));
        size_t o = (size_t)(mbase + j) * N + col;
        if (res)  v += res[o];
        if (outf) outf[o] = v;
        if (outb) outb[o] = f2bf_u(v);
    }
}

__global__ __launch_bounds__(256) void gemm_bf16_k(
    const unsigned short* __restrict__ A,   // [M,K] bf16
    const unsigned short* __restrict__ W,   // [N,K] bf16 (PyTorch Linear layout)
    const float* __restrict__ bias,         // [N] or null
    const float* __restrict__ res,          // [M,N] or null (added after act)
    float* __restrict__ outf,               // [M,N] or null
    unsigned short* __restrict__ outb,      // [M,N] bf16 or null
    int M, int N, int K, int act)
{
    extern __shared__ unsigned short wtile[];    // [2][256*64] bf16 = 2 x 32KB
    const int lane  = threadIdx.x & 31;
    const int wave  = threadIdx.x >> 5;
    const int wm    = wave >> 2;                 // 0..1
    const int wn    = wave & 3;                  // 0..3
    const int m0    = blockIdx.y * 32 + wm * 16;
    const int nblk  = blockIdx.x * 256;
    const int ln    = lane & 15;
    const int kba   = (lane >> 4) << 4;          // A frag: 0 or 16
    const int kbb   = (lane >> 4) << 4;          // B frag: 0 or 16

    v8f acc0 = {}, acc1 = {}, acc2 = {}, acc3 = {};
    const unsigned short* arow = A + (size_t)(m0 + ln) * K;
    const unsigned short* wblk = W + (size_t)nblk * K;
    const int nchunks = K >> 6;                  // K / 64

    // Prologue: TDM stage chunk 0 into buffer 0
    if (threadIdx.x < 32) {
        tdm_load_tile_2d(0u, wblk, 256, 64, K);
        __builtin_amdgcn_s_wait_tensorcnt(0);
    }
    __syncthreads();

    for (int kc = 0; kc < nchunks; ++kc) {
        const int buf = kc & 1;
        if (threadIdx.x < 32 && (kc + 1) < nchunks) {
            tdm_load_tile_2d((unsigned)((1 - buf) * (256 * 64 * 2)),
                             wblk + (size_t)(kc + 1) * 64, 256, 64, K);
        }
        const unsigned short* bt = wtile + buf * (256 * 64);
        const unsigned short* b0r = bt + (wn * 64 +  0 + ln) * 64 + kbb;
        const unsigned short* b1r = bt + (wn * 64 + 16 + ln) * 64 + kbb;
        const unsigned short* b2r = bt + (wn * 64 + 32 + ln) * 64 + kbb;
        const unsigned short* b3r = bt + (wn * 64 + 48 + ln) * 64 + kbb;
#pragma unroll
        for (int ks = 0; ks < 2; ++ks) {
            const int kg = kc * 64 + ks * 32;    // global K offset
            const int kl = ks * 32;              // K offset within LDS tile row
            BF16Frag a;
            a.q[0] = *(const u32x4*)(arow + kg + kba);
            a.q[1] = *(const u32x4*)(arow + kg + kba + 8);
            BF16Frag b0, b1, b2, b3;
            b0.q[0] = *(const u32x4*)(b0r + kl);  b0.q[1] = *(const u32x4*)(b0r + kl + 8);
            b1.q[0] = *(const u32x4*)(b1r + kl);  b1.q[1] = *(const u32x4*)(b1r + kl + 8);
            b2.q[0] = *(const u32x4*)(b2r + kl);  b2.q[1] = *(const u32x4*)(b2r + kl + 8);
            b3.q[0] = *(const u32x4*)(b3r + kl);  b3.q[1] = *(const u32x4*)(b3r + kl + 8);
            acc0 = WMMA_BF16(a.v, b0.v, acc0);
            acc1 = WMMA_BF16(a.v, b1.v, acc1);
            acc2 = WMMA_BF16(a.v, b2.v, acc2);
            acc3 = WMMA_BF16(a.v, b3.v, acc3);
        }
        // Drain the in-flight TDM before anyone touches the other buffer.
        if (threadIdx.x < 32) __builtin_amdgcn_s_wait_tensorcnt(0);
        __syncthreads();
    }

    const int mbase = m0 + ((lane >> 4) << 3);
    const int n0 = nblk + wn * 64;
    gemm_epilogue(acc0, mbase, n0 +  0 + ln, N, bias, res, outf, outb, act);
    gemm_epilogue(acc1, mbase, n0 + 16 + ln, N, bias, res, outf, outb, act);
    gemm_epilogue(acc2, mbase, n0 + 32 + ln, N, bias, res, outf, outb, act);
    gemm_epilogue(acc3, mbase, n0 + 48 + ln, N, bias, res, outf, outb, act);
}

// ---------------------------------------------------------------------------
// GRU step: hh = h_in @ W_hh^T, fused gate pointwise, double-buffered state.
// Grid 12 x 128 threads (4 waves). Each wave owns one 16-col c-tile and its
// three gate row-blocks of W_hh (r, z, n) so the combine stays in-register.
// M = 8 (padded to 16 with zero rows). Prefetches next timestep's xi.
// ---------------------------------------------------------------------------
__global__ __launch_bounds__(128) void gru_step_k(
    const unsigned short* __restrict__ h_in_b,  // [8,768] bf16
    const float* __restrict__ h_in_f,           // [8,768] f32
    unsigned short* __restrict__ h_out_b,
    float* __restrict__ h_out_f,
    const unsigned short* __restrict__ Whh_b,   // [2304,768] bf16
    const float* __restrict__ b_hh,             // [2304]
    const float* __restrict__ xi,               // [8,2048,2304] (b_ih included)
    float* __restrict__ h_rnn,                  // [8,2048,768]
    unsigned short* __restrict__ h_rnn_b,
    int t)
{
    const int lane = threadIdx.x & 31;
    const int wave = threadIdx.x >> 5;
    const int tile = blockIdx.x * 4 + wave;      // 0..47
    const int c0   = tile * 16;
    const int ln   = lane & 15;
    const int kba  = (lane >> 4) << 4;
    const int kbb  = (lane >> 4) << 4;
    const u32x4 zq = {0u, 0u, 0u, 0u};

    const int am = ln;                           // A row (batch), valid if < 8
    const bool mval = (am < 8);

    // Prefetch next timestep's gate inputs for this wave's columns.
    if (mval && (t + 1) < TT) {
        const float* pf = xi + ((size_t)am * TT + (t + 1)) * (3*CC) + c0 + ln;
        __builtin_prefetch(pf, 0, 1);
        __builtin_prefetch(pf + CC, 0, 1);
        __builtin_prefetch(pf + 2*CC, 0, 1);
    }

    v8f ar = {}, az = {}, an = {};
    const unsigned short* wr = Whh_b + (size_t)(c0 + ln) * CC;
    const unsigned short* wz = wr + (size_t)CC * CC;
    const unsigned short* wn = wz + (size_t)CC * CC;
    const unsigned short* hrow = h_in_b + (size_t)am * CC;

    for (int k0 = 0; k0 < CC; k0 += 32) {
        BF16Frag a;
        if (mval) {
            a.q[0] = *(const u32x4*)(hrow + k0 + kba);
            a.q[1] = *(const u32x4*)(hrow + k0 + kba + 8);
        } else { a.q[0] = zq; a.q[1] = zq; }
        BF16Frag br, bz, bn;
        br.q[0] = *(const u32x4*)(wr + k0 + kbb);  br.q[1] = *(const u32x4*)(wr + k0 + kbb + 8);
        bz.q[0] = *(const u32x4*)(wz + k0 + kbb);  bz.q[1] = *(const u32x4*)(wz + k0 + kbb + 8);
        bn.q[0] = *(const u32x4*)(wn + k0 + kbb);  bn.q[1] = *(const u32x4*)(wn + k0 + kbb + 8);
        ar = WMMA_BF16(a.v, br.v, ar);
        az = WMMA_BF16(a.v, bz.v, az);
        an = WMMA_BF16(a.v, bn.v, an);
    }

    const int half = (lane >> 4) << 3;
    const int col  = c0 + ln;
#pragma unroll
    for (int j = 0; j < 8; ++j) {
        int m = j + half;                        // batch index
        if (m < 8) {
            size_t xo = ((size_t)m * TT + t) * (3*CC) + col;
            float xr = xi[xo], xz = xi[xo + CC], xn = xi[xo + 2*CC];
            float hr = ar[j] + b_hh[col];
            float hz = az[j] + b_hh[CC + col];
            float hn = an[j] + b_hh[2*CC + col];
            float r  = 1.0f / (1.0f + expf(-(xr + hr)));
            float z  = 1.0f / (1.0f + expf(-(xz + hz)));
            float nn = tanhf(xn + r * hn);
            float hp = h_in_f[(size_t)m * CC + col];
            float hv = (1.0f - z) * nn + z * hp;
            h_out_f[(size_t)m * CC + col] = hv;
            h_out_b[(size_t)m * CC + col] = f2bf_u(hv);
            size_t ho = ((size_t)m * TT + t) * CC + col;
            h_rnn[ho]   = hv;
            h_rnn_b[ho] = f2bf_u(hv);
        }
    }
}

// ---------------------------------------------------------------------------
// Gate MLP: energy = sigmoid(tanh(h @ w1^T + b1) @ w2^T + b2)
// One wave per token; lane j owns hidden unit j of 32.
// ---------------------------------------------------------------------------
__global__ __launch_bounds__(32) void gate_mlp_k(
    const float* __restrict__ h_rnn, const float* __restrict__ w1,
    const float* __restrict__ b1,    const float* __restrict__ w2,
    const float* __restrict__ b2,    float* __restrict__ energy)
{
    int tok = blockIdx.x, j = threadIdx.x;
    const float* hv = h_rnn + (size_t)tok * CC;
    const float* wr = w1 + (size_t)j * CC;
    float acc = b1[j];
    for (int c = 0; c < CC; ++c) acc += hv[c] * wr[c];
    float tv = tanhf(acc) * w2[j];
    for (int d = 16; d > 0; d >>= 1) tv += __shfl_down(tv, d, 32);
    if (j == 0) energy[tok] = 1.0f / (1.0f + expf(-(tv + b2[0])));
}

// ---------------------------------------------------------------------------
// Stable top-k (k=1024 of 2048) by exact rank counting + in-order compaction.
// One 256-thread block per batch.
// ---------------------------------------------------------------------------
__global__ __launch_bounds__(256) void topk_select_k(
    const float* __restrict__ energy, int* __restrict__ idx)
{
    __shared__ float e[TT];
    __shared__ int cnts[256];
    __shared__ int offs[256];
    int b = blockIdx.x, tid = threadIdx.x;
    for (int i = tid; i < TT; i += 256) e[i] = energy[b * TT + i];
    __syncthreads();
    int flags = 0, nsel = 0;
    int i0 = tid * 8;
#pragma unroll 1
    for (int u = 0; u < 8; ++u) {
        int i = i0 + u; float ei = e[i]; int cnt = 0;
        for (int jx = 0; jx < TT; ++jx) {
            float ej = e[jx];
            cnt += (ej > ei) || (ej == ei && jx < i);
        }
        if (cnt < KSEL) { flags |= (1 << u); ++nsel; }
    }
    cnts[tid] = nsel;
    __syncthreads();
    if (tid == 0) { int s = 0; for (int i = 0; i < 256; ++i) { offs[i] = s; s += cnts[i]; } }
    __syncthreads();
    int pos = offs[tid];
    for (int u = 0; u < 8; ++u)
        if (flags & (1 << u)) idx[b * KSEL + (pos++)] = i0 + u;
}

// ---------------------------------------------------------------------------
// Gather selected tokens (bf16) for the qkv GEMM
// ---------------------------------------------------------------------------
__global__ __launch_bounds__(256) void gather_k(
    const unsigned short* __restrict__ h_b, const int* __restrict__ idx,
    unsigned short* __restrict__ xsel)
{
    int i = blockIdx.x * 256 + threadIdx.x;      // MS*CC total
    int c = i % CC; int p = (i / CC) % KSEL; int b = i / (CC * KSEL);
    int t = idx[b * KSEL + p];
    xsel[i] = h_b[((size_t)(b * TT + t)) * CC + c];
}

// ---------------------------------------------------------------------------
// Flash-style causal attention over selected tokens. One wave per 16-query
// tile per (b,h). QK^T and P@V both via bf16 WMMA; P transposed through a
// per-wave LDS tile (per-wave DS ordering + wave_barrier).
// ---------------------------------------------------------------------------
__global__ __launch_bounds__(256) void attention_k(
    const unsigned short* __restrict__ qkv,   // [B,1024,2304] bf16 (q|k|v)
    unsigned short* __restrict__ yout)        // [B,1024,768] bf16
{
    __shared__ float ptile[8][256];
    const int lane = threadIdx.x & 31;
    const int wave = threadIdx.x >> 5;
    const int gw   = blockIdx.x * 8 + wave;   // 0..6143
    const int qt   = gw & 63;
    const int tmp  = gw >> 6;
    const int hh   = tmp % HH;
    const int bb   = tmp / HH;
    const int q0   = qt * 16;
    const int ln   = lane & 15;
    const int kba  = (lane >> 4) << 4;
    const int kbb  = (lane >> 4) << 4;
    const int half = (lane >> 4) << 3;
    volatile float* pt = &ptile[wave][0];

    const unsigned short* qrow = qkv + ((size_t)(bb * KSEL + q0 + ln)) * (3*CC) + hh * HD;
    BF16Frag qa0, qa1;
    qa0.q[0] = *(const u32x4*)(qrow + kba);
    qa0.q[1] = *(const u32x4*)(qrow + kba + 8);
    qa1.q[0] = *(const u32x4*)(qrow + 32 + kba);
    qa1.q[1] = *(const u32x4*)(qrow + 32 + kba + 8);

    v8f y0 = {}, y1 = {}, y2 = {}, y3 = {};
    float rmax[8], rsum[8];
#pragma unroll
    for (int j = 0; j < 8; ++j) { rmax[j] = -3.0e38f; rsum[j] = 0.0f; }

    for (int kt = 0; kt <= qt; ++kt) {
        const unsigned short* krow =
            qkv + ((size_t)(bb * KSEL + kt * 16 + ln)) * (3*CC) + CC + hh * HD;
        BF16Frag kf0, kf1;
        kf0.q[0] = *(const u32x4*)(krow + kbb);
        kf0.q[1] = *(const u32x4*)(krow + kbb + 8);
        kf1.q[0] = *(const u32x4*)(krow + 32 + kbb);
        kf1.q[1] = *(const u32x4*)(krow + 32 + kbb + 8);
        v8f s = {};
        s = WMMA_BF16(qa0.v, kf0.v, s);
        s = WMMA_BF16(qa1.v, kf1.v, s);

        const int kg = kt * 16 + ln;
        float p[8];
#pragma unroll
        for (int j = 0; j < 8; ++j) {
            int m = j + half;
            float sv = s[j] * 0.125f;                 // 1/sqrt(64)
            if (kg > q0 + m) sv = -1.0e30f;           // causal mask
            float mx = sv;
            mx = fmaxf(mx, __shfl_xor(mx, 1, 32));
            mx = fmaxf(mx, __shfl_xor(mx, 2, 32));
            mx = fmaxf(mx, __shfl_xor(mx, 4, 32));
            mx = fmaxf(mx, __shfl_xor(mx, 8, 32));
            float nm = fmaxf(rmax[j], mx);
            float sc = expf(rmax[j] - nm);
            rmax[j] = nm;
            float pv = expf(sv - nm);
            float ps = pv;
            ps += __shfl_xor(ps, 1, 32);
            ps += __shfl_xor(ps, 2, 32);
            ps += __shfl_xor(ps, 4, 32);
            ps += __shfl_xor(ps, 8, 32);
            rsum[j] = rsum[j] * sc + ps;
            y0[j] *= sc; y1[j] *= sc; y2[j] *= sc; y3[j] *= sc;
            p[j] = pv;
        }
        // P (C-layout) -> LDS -> A-layout bf16 frag, K padded 16..31 with zero
#pragma unroll
        for (int j = 0; j < 8; ++j) pt[(j + half) * 16 + ln] = p[j];
        __builtin_amdgcn_wave_barrier();
        BF16Frag pa;
#pragma unroll
        for (int i = 0; i < 16; ++i) pa.h[i] = kba ? (unsigned short)0 : f2bf_u(pt[ln * 16 + i]);
        __builtin_amdgcn_wave_barrier();

        const size_t vbase = ((size_t)(bb * KSEL + kt * 16)) * (3*CC) + 2*CC + hh * HD;
#pragma unroll
        for (int c = 0; c < 4; ++c) {
            BF16Frag vb;
            if (lane < 16) {
#pragma unroll
                for (int i = 0; i < 16; ++i)
                    vb.h[i] = qkv[vbase + (size_t)i * (3*CC) + c * 16 + ln];
            } else {
#pragma unroll
                for (int i = 0; i < 16; ++i) vb.h[i] = 0;
            }
            if      (c == 0) y0 = WMMA_BF16(pa.v, vb.v, y0);
            else if (c == 1) y1 = WMMA_BF16(pa.v, vb.v, y1);
            else if (c == 2) y2 = WMMA_BF16(pa.v, vb.v, y2);
            else             y3 = WMMA_BF16(pa.v, vb.v, y3);
        }
    }
#pragma unroll
    for (int c = 0; c < 4; ++c) {
        v8f yy = (c == 0) ? y0 : ((c == 1) ? y1 : ((c == 2) ? y2 : y3));
#pragma unroll
        for (int j = 0; j < 8; ++j) {
            int m = j + half;
            float val = yy[j] / rsum[j];
            yout[((size_t)(bb * KSEL + q0 + m)) * CC + hh * HD + c * 16 + ln] = f2bf_u(val);
        }
    }
}

// ---------------------------------------------------------------------------
// Scatter weighted attention output back to full sequence
// ---------------------------------------------------------------------------
__global__ __launch_bounds__(256) void scatter_k(
    const float* __restrict__ proj, const int* __restrict__ idx,
    const float* __restrict__ energy, float* __restrict__ attn)
{
    int i = blockIdx.x * 256 + threadIdx.x;      // MS*CC total
    int c = i % CC; int p = (i / CC) % KSEL; int b = i / (CC * KSEL);
    int t = idx[b * KSEL + p];
    attn[((size_t)(b * TT + t)) * CC + c] = proj[i] * energy[b * TT + t];
}

// ---------------------------------------------------------------------------
// LayerNorm(h_rnn + attn) -> f32 + bf16
// ---------------------------------------------------------------------------
__global__ __launch_bounds__(256) void layernorm_k(
    const float* __restrict__ h_rnn, const float* __restrict__ attn,
    const float* __restrict__ g, const float* __restrict__ bln,
    float* __restrict__ outf, unsigned short* __restrict__ outb)
{
    __shared__ float red[16];
    int tok = blockIdx.x, tid = threadIdx.x;
    const float* a = h_rnn + (size_t)tok * CC;
    const float* c = attn  + (size_t)tok * CC;
    float v0 = a[tid]       + c[tid];
    float v1 = a[tid + 256] + c[tid + 256];
    float v2 = a[tid + 512] + c[tid + 512];
    float s  = v0 + v1 + v2;
    float s2 = v0 * v0 + v1 * v1 + v2 * v2;
    for (int d = 16; d > 0; d >>= 1) { s += __shfl_down(s, d, 32); s2 += __shfl_down(s2, d, 32); }
    if ((tid & 31) == 0) { red[tid >> 5] = s; red[8 + (tid >> 5)] = s2; }
    __syncthreads();
    if (tid == 0) {
        float ts = 0, ts2 = 0;
        for (int i = 0; i < 8; ++i) { ts += red[i]; ts2 += red[8 + i]; }
        red[0] = ts; red[1] = ts2;
    }
    __syncthreads();
    float mu  = red[0] * (1.0f / CC);
    float var = red[1] * (1.0f / CC) - mu * mu;
    float inv = rsqrtf(var + 1e-5f);
    size_t o = (size_t)tok * CC;
#pragma unroll
    for (int u = 0; u < 3; ++u) {
        int col = tid + u * 256;
        float v = (u == 0) ? v0 : ((u == 1) ? v1 : v2);
        float r = (v - mu) * inv * g[col] + bln[col];
        outf[o + col] = r;
        outb[o + col] = f2bf_u(r);
    }
}

// ---------------------------------------------------------------------------
// Copy energy to d_out + active_count reduction (single block)
// ---------------------------------------------------------------------------
__global__ __launch_bounds__(256) void finalize_k(
    const float* __restrict__ energy, float* __restrict__ dout)
{
    __shared__ float red[8];
    int tid = threadIdx.x;
    float cnt = 0.0f;
    for (int i = tid; i < MT; i += 256) {
        float ev = energy[i];
        dout[HOUT + i] = ev;
        cnt += (ev > 0.5f) ? 1.0f : 0.0f;
    }
    for (int d = 16; d > 0; d >>= 1) cnt += __shfl_down(cnt, d, 32);
    if ((tid & 31) == 0) red[tid >> 5] = cnt;
    __syncthreads();
    if (tid == 0) {
        float s = 0; for (int i = 0; i < 8; ++i) s += red[i];
        dout[HOUT + MT] = s;
    }
}

// ---------------------------------------------------------------------------
// Host orchestration
// ---------------------------------------------------------------------------
extern "C" void kernel_launch(void* const* d_in, const int* in_sizes, int n_in,
                              void* d_out, int out_size, void* d_ws, size_t ws_size,
                              hipStream_t stream) {
    const float* x     = (const float*)d_in[0];
    const float* W_ih  = (const float*)d_in[1];
    const float* W_hh  = (const float*)d_in[2];
    const float* b_ih  = (const float*)d_in[3];
    const float* b_hh  = (const float*)d_in[4];
    const float* gw1   = (const float*)d_in[5];
    const float* gb1   = (const float*)d_in[6];
    const float* gw2   = (const float*)d_in[7];
    const float* gb2   = (const float*)d_in[8];
    const float* qkvw  = (const float*)d_in[9];
    const float* qkvb  = (const float*)d_in[10];
    const float* projw = (const float*)d_in[11];
    const float* projb = (const float*)d_in[12];
    const float* lng   = (const float*)d_in[13];
    const float* lnb   = (const float*)d_in[14];
    const float* f1w   = (const float*)d_in[15];
    const float* f1b   = (const float*)d_in[16];
    const float* f2w   = (const float*)d_in[17];
    const float* f2b   = (const float*)d_in[18];
    float* out = (float*)d_out;

    char* ws = (char*)d_ws;
    size_t off = 0;
    auto alloc = [&](size_t bytes) -> char* {
        char* p = ws + off; off += (bytes + 255) & ~(size_t)255; return p;
    };

    unsigned short* x_b     = (unsigned short*)alloc((size_t)MT * CC * 2);
    unsigned short* wih_b   = (unsigned short*)alloc((size_t)3*CC * CC * 2);
    unsigned short* whh_b   = (unsigned short*)alloc((size_t)3*CC * CC * 2);
    unsigned short* qkvw_b  = (unsigned short*)alloc((size_t)3*CC * CC * 2);
    unsigned short* projw_b = (unsigned short*)alloc((size_t)CC * CC * 2);
    unsigned short* f1w_b   = (unsigned short*)alloc((size_t)DFF * CC * 2);
    unsigned short* f2w_b   = (unsigned short*)alloc((size_t)CC * DFF * 2);
    float*          xi      = (float*)alloc((size_t)MT * 3*CC * 4);
    float*          h_rnn   = (float*)alloc((size_t)MT * CC * 4);
    unsigned short* h_rnn_b = (unsigned short*)alloc((size_t)MT * CC * 2);
    float*          hs_f0   = (float*)alloc((size_t)BB * CC * 4);
    float*          hs_f1   = (float*)alloc((size_t)BB * CC * 4);
    unsigned short* hs_b0   = (unsigned short*)alloc((size_t)BB * CC * 2);
    unsigned short* hs_b1   = (unsigned short*)alloc((size_t)BB * CC * 2);
    float*          energy  = (float*)alloc((size_t)MT * 4);
    int*            idx     = (int*)alloc((size_t)BB * KSEL * 4);
    unsigned short* xsel_b  = (unsigned short*)alloc((size_t)MS * CC * 2);
    unsigned short* qkv_b   = (unsigned short*)alloc((size_t)MS * 3*CC * 2);
    unsigned short* y_b     = (unsigned short*)alloc((size_t)MS * CC * 2);
    float*          proj_f  = (float*)alloc((size_t)MS * CC * 4);
    float*          attn_f  = (float*)alloc((size_t)MT * CC * 4);
    float*          hln_f   = (float*)alloc((size_t)MT * CC * 4);
    unsigned short* hln_b   = (unsigned short*)alloc((size_t)MT * CC * 2);
    unsigned short* act_b   = (unsigned short*)alloc((size_t)MT * DFF * 2);

    auto cvt = [&](const float* src, unsigned short* dst, int n) {
        f32_to_bf16_k<<<(n + 255) / 256, 256, 0, stream>>>(src, dst, n);
    };
    cvt(x,     x_b,     MT * CC);
    cvt(W_ih,  wih_b,   3*CC * CC);
    cvt(W_hh,  whh_b,   3*CC * CC);
    cvt(qkvw,  qkvw_b,  3*CC * CC);
    cvt(projw, projw_b, CC * CC);
    cvt(f1w,   f1w_b,   DFF * CC);
    cvt(f2w,   f2w_b,   CC * DFF);

    (void)hipMemsetAsync(hs_f0, 0, (size_t)BB * CC * 4, stream);
    (void)hipMemsetAsync(hs_b0, 0, (size_t)BB * CC * 2, stream);
    (void)hipMemsetAsync(attn_f, 0, (size_t)MT * CC * 4, stream);

    const size_t GEMM_LDS = 2 * 256 * 64 * 2;   // 64 KB double-buffered weight tile

    // xi = x @ W_ih^T + b_ih   [16384, 2304]
    gemm_bf16_k<<<dim3(3*CC/256, MT/32), 256, GEMM_LDS, stream>>>(
        x_b, wih_b, b_ih, nullptr, xi, nullptr, MT, 3*CC, CC, 0);

    // GRU scan (sequential, double-buffered state)
    for (int t = 0; t < TT; ++t) {
        const unsigned short* hib = (t & 1) ? hs_b1 : hs_b0;
        const float*          hif = (t & 1) ? hs_f1 : hs_f0;
        unsigned short*       hob = (t & 1) ? hs_b0 : hs_b1;
        float*                hof = (t & 1) ? hs_f0 : hs_f1;
        gru_step_k<<<12, 128, 0, stream>>>(hib, hif, hob, hof, whh_b, b_hh,
                                           xi, h_rnn, h_rnn_b, t);
    }

    gate_mlp_k<<<MT, 32, 0, stream>>>(h_rnn, gw1, gb1, gw2, gb2, energy);
    topk_select_k<<<BB, 256, 0, stream>>>(energy, idx);
    gather_k<<<(MS * CC) / 256, 256, 0, stream>>>(h_rnn_b, idx, xsel_b);

    // qkv = x_sel @ qkv_w^T + qkv_b   [8192, 2304] -> bf16
    gemm_bf16_k<<<dim3(3*CC/256, MS/32), 256, GEMM_LDS, stream>>>(
        xsel_b, qkvw_b, qkvb, nullptr, nullptr, qkv_b, MS, 3*CC, CC, 0);

    attention_k<<<(BB * HH * (KSEL/16)) / 8, 256, 0, stream>>>(qkv_b, y_b);

    // proj = y @ proj_w^T + proj_b   [8192, 768]
    gemm_bf16_k<<<dim3(CC/256, MS/32), 256, GEMM_LDS, stream>>>(
        y_b, projw_b, projb, nullptr, proj_f, nullptr, MS, CC, CC, 0);

    scatter_k<<<(MS * CC) / 256, 256, 0, stream>>>(proj_f, idx, energy, attn_f);
    layernorm_k<<<MT, 256, 0, stream>>>(h_rnn, attn_f, lng, lnb, hln_f, hln_b);

    // ffn1 = GELU(h_ln @ ffn_w1^T + b1)   [16384, 3072] -> bf16
    gemm_bf16_k<<<dim3(DFF/256, MT/32), 256, GEMM_LDS, stream>>>(
        hln_b, f1w_b, f1b, nullptr, nullptr, act_b, MT, DFF, CC, 1);

    // out_h = h_ln + (act @ ffn_w2^T + b2)   [16384, 768] -> d_out
    gemm_bf16_k<<<dim3(CC/256, MT/32), 256, GEMM_LDS, stream>>>(
        act_b, f2w_b, f2b, hln_f, out, nullptr, MT, CC, DFF, 0);

    finalize_k<<<1, 256, 0, stream>>>(energy, out);
    (void)in_sizes; (void)n_in; (void)out_size; (void)ws_size; (void)gb2;
}
